// SlidingWindowMultiheadAttention_40630390620258
// MI455X (gfx1250) — matrix operand order repaired and checked by reading, not compiled, not run on hardware
//
#include <hip/hip_runtime.h>

#define D_MODEL   1024
#define NUM_HEADS 16
#define HEAD_DIM  64
#define WINDOW    256
#define BATCH     2
#define SEQ       2048
#define M_TOTAL   (BATCH * SEQ)   // 4096

typedef __attribute__((ext_vector_type(16))) __bf16 v16bf;
typedef __attribute__((ext_vector_type(8)))  float  v8f;

union FragBF { v16bf v; unsigned int u[8]; unsigned short h[16]; };
union FragF  { v8f v; float f[8]; };

__device__ __forceinline__ unsigned short f32_to_bf16(float x) {
  union { float f; unsigned int u; } c; c.f = x;
  unsigned int r = c.u + 0x7FFFu + ((c.u >> 16) & 1u);  // round-to-nearest-even
  return (unsigned short)(r >> 16);
}

// ---------------------------------------------------------------------------
// CDNA5 async global->LDS copy (ASYNCcnt path, §15.18 / CDNA5 ISA).
// VDST = LDS byte address (low 32 bits of the generic pointer: the LDS
// aperture base has zero low bits), VADDR = 64-bit global address, SADDR=off.
// ---------------------------------------------------------------------------
__device__ __forceinline__ void async_copy_b128(unsigned short* dstLds,
                                                const unsigned short* srcGlobal) {
  unsigned int la = (unsigned int)(unsigned long long)(void*)dstLds;
  unsigned long long ga = (unsigned long long)(const void*)srcGlobal;
  asm volatile("global_load_async_to_lds_b128 %0, %1, off"
               :: "v"(la), "v"(ga) : "memory");
}
__device__ __forceinline__ void wait_async0() {
  asm volatile("s_wait_asynccnt 0x0" ::: "memory");
}

// ---------------------------------------------------------------------------
// fp32 -> bf16 bulk conversion (8 elems / thread, n is a multiple of 8)
// ---------------------------------------------------------------------------
__global__ void cvt_f32_bf16(const float* __restrict__ in,
                             unsigned short* __restrict__ out, int n) {
  int i = (blockIdx.x * blockDim.x + threadIdx.x) * 8;
  if (i >= n) return;
  float4 a = *(const float4*)(in + i);
  float4 b = *(const float4*)(in + i + 4);
  uint4 o;
  o.x = (unsigned int)f32_to_bf16(a.x) | ((unsigned int)f32_to_bf16(a.y) << 16);
  o.y = (unsigned int)f32_to_bf16(a.z) | ((unsigned int)f32_to_bf16(a.w) << 16);
  o.z = (unsigned int)f32_to_bf16(b.x) | ((unsigned int)f32_to_bf16(b.y) << 16);
  o.w = (unsigned int)f32_to_bf16(b.z) | ((unsigned int)f32_to_bf16(b.w) << 16);
  *(uint4*)(out + i) = o;
}

// ---------------------------------------------------------------------------
// Shared WMMA GEMM mainloop: acc += A[M,K] * W[N,K]^T, bf16 in / f32 acc.
// Block tile 128x128, 8 waves (2x4) of 64x32, K-step 32.
// Double-buffered LDS fed by async global->LDS copies.
// ---------------------------------------------------------------------------
#define BM 128
#define BN 128
#define BK 32
#define LDSP 40   // padded row stride (ushorts); 80B keeps 16B alignment

__device__ __forceinline__ void gemm_core(const unsigned short* __restrict__ Ag,
                                          const unsigned short* __restrict__ Bg,
                                          unsigned short (&As)[2][BM][LDSP],
                                          unsigned short (&Bs)[2][BM][LDSP],
                                          int m0, int n0, FragF (&acc)[4][2]) {
  const int tid  = threadIdx.x;
  const int lane = tid & 31;
  const int wid  = tid >> 5;
  const int wm   = wid >> 2;
  const int wn   = wid & 3;
  const int lh   = lane >> 4;
  const int l16  = lane & 15;
  const int lr   = tid >> 1;        // 0..127: staged row
  const int lc   = (tid & 1) * 16;  // 0/16:   staged 16-elem chunk

  const unsigned short* ga = Ag + (size_t)(m0 + lr) * D_MODEL + lc;
  const unsigned short* gb = Bg + (size_t)(n0 + lr) * D_MODEL + lc;

  // prologue: stage k-tile 0 into buffer 0
  async_copy_b128(&As[0][lr][lc],     ga);
  async_copy_b128(&As[0][lr][lc + 8], ga + 8);
  async_copy_b128(&Bs[0][lr][lc],     gb);
  async_copy_b128(&Bs[0][lr][lc + 8], gb + 8);

  const int NK = D_MODEL / BK;  // 32
  for (int i = 0; i < NK; ++i) {
    const int cur = i & 1;
    wait_async0();       // this wave's stage of buffer `cur` landed in LDS
    __syncthreads();     // everyone's stage of `cur` visible

    if (i + 1 < NK) {    // overlap next stage with this tile's WMMAs
      const int k = (i + 1) * BK;
      async_copy_b128(&As[cur ^ 1][lr][lc],     ga + k);
      async_copy_b128(&As[cur ^ 1][lr][lc + 8], ga + k + 8);
      async_copy_b128(&Bs[cur ^ 1][lr][lc],     gb + k);
      async_copy_b128(&Bs[cur ^ 1][lr][lc + 8], gb + k + 8);
    }

    FragBF bfr[2];
#pragma unroll
    for (int fn = 0; fn < 2; ++fn) {
      const int bn = wn * 32 + fn * 16 + l16;
#pragma unroll
      for (int j = 0; j < 8; ++j)
        bfr[fn].u[j] = *(const unsigned int*)&Bs[cur][bn][16 * lh + 2 * j];
    }
#pragma unroll
    for (int fm = 0; fm < 4; ++fm) {
      FragBF afr;
      const int am = wm * 64 + fm * 16 + l16;
#pragma unroll
      for (int j = 0; j < 8; ++j) {
        const int kk = 16 * lh + 2 * j;
        afr.u[j] = *(const unsigned int*)&As[cur][am][kk];
      }
#pragma unroll
      for (int fn = 0; fn < 2; ++fn)
        acc[fm][fn].v = __builtin_amdgcn_wmma_f32_16x16x32_bf16(
            false, afr.v, false, bfr[fn].v, (short)0, acc[fm][fn].v, false, false);
    }
    __syncthreads();     // all reads of `cur` done before it is restaged
  }
}

// ---------------------------------------------------------------------------
// QKV projection; gridDim.z selects Wq/Wk/Wv. Epilogue layouts:
//   z==0: Q  -> [B,H,S,64] bf16     z==1: K -> [B,H,S,64] bf16
//   z==2: Vt -> [B,H,64,S] bf16 (transposed; packed b128 stores along S)
// ---------------------------------------------------------------------------
__launch_bounds__(256)
__global__ void gemm_qkv_kernel(const unsigned short* __restrict__ Xb,
                                const unsigned short* __restrict__ Wq,
                                const unsigned short* __restrict__ Wk,
                                const unsigned short* __restrict__ Wv,
                                unsigned short* __restrict__ Qo,
                                unsigned short* __restrict__ Ko,
                                unsigned short* __restrict__ Vt) {
  const int which = blockIdx.z;
  const unsigned short* __restrict__ W =
      (which == 0) ? Wq : ((which == 1) ? Wk : Wv);
  const int m0 = blockIdx.y * BM;
  const int n0 = blockIdx.x * BN;

  __shared__ __align__(16) unsigned short As[2][BM][LDSP];
  __shared__ __align__(16) unsigned short Bs[2][BM][LDSP];

  const int lane = threadIdx.x & 31;
  const int wid  = threadIdx.x >> 5;
  const int wm = wid >> 2, wn = wid & 3;
  const int lh = lane >> 4, l16 = lane & 15;

  FragF acc[4][2];
#pragma unroll
  for (int i = 0; i < 4; ++i)
#pragma unroll
    for (int j = 0; j < 2; ++j)
#pragma unroll
      for (int r = 0; r < 8; ++r) acc[i][j].f[r] = 0.0f;

  gemm_core(Xb, W, As, Bs, m0, n0, acc);

  if (which == 2) {
    // Vt[b][h][d][s]: r=0..7 are consecutive s at fixed d -> one b128 store
#pragma unroll
    for (int fm = 0; fm < 4; ++fm)
#pragma unroll
      for (int fn = 0; fn < 2; ++fn) {
        const int mb = m0 + wm * 64 + fm * 16 + 8 * lh;  // row of r==0
        const int n  = n0 + wn * 32 + fn * 16 + l16;
        const int bb = mb >> 11, s = mb & 2047;
        const int hh = n >> 6,   dd = n & 63;
        uint4 pk;
        pk.x = (unsigned int)f32_to_bf16(acc[fm][fn].f[0]) |
               ((unsigned int)f32_to_bf16(acc[fm][fn].f[1]) << 16);
        pk.y = (unsigned int)f32_to_bf16(acc[fm][fn].f[2]) |
               ((unsigned int)f32_to_bf16(acc[fm][fn].f[3]) << 16);
        pk.z = (unsigned int)f32_to_bf16(acc[fm][fn].f[4]) |
               ((unsigned int)f32_to_bf16(acc[fm][fn].f[5]) << 16);
        pk.w = (unsigned int)f32_to_bf16(acc[fm][fn].f[6]) |
               ((unsigned int)f32_to_bf16(acc[fm][fn].f[7]) << 16);
        *(uint4*)&Vt[(((size_t)(bb * NUM_HEADS + hh) * HEAD_DIM) + dd) * SEQ + s] = pk;
      }
  } else {
    unsigned short* __restrict__ O = (which == 0) ? Qo : Ko;
#pragma unroll
    for (int fm = 0; fm < 4; ++fm)
#pragma unroll
      for (int fn = 0; fn < 2; ++fn)
#pragma unroll
        for (int r = 0; r < 8; ++r) {
          const int m = m0 + wm * 64 + fm * 16 + r + 8 * lh;
          const int n = n0 + wn * 32 + fn * 16 + l16;
          const int bb = m >> 11, s = m & 2047;
          const int hh = n >> 6,  dd = n & 63;
          O[(((size_t)(bb * NUM_HEADS + hh) * SEQ) + s) * HEAD_DIM + dd] =
              f32_to_bf16(acc[fm][fn].f[r]);
        }
  }
}

// ---------------------------------------------------------------------------
// Output projection: out[M,N] f32 = A[M,K]bf16 * Wo[N,K]^T
// ---------------------------------------------------------------------------
__launch_bounds__(256)
__global__ void gemm_out_kernel(const unsigned short* __restrict__ Ab,
                                const unsigned short* __restrict__ W,
                                float* __restrict__ out) {
  const int m0 = blockIdx.y * BM;
  const int n0 = blockIdx.x * BN;

  __shared__ __align__(16) unsigned short As[2][BM][LDSP];
  __shared__ __align__(16) unsigned short Bs[2][BM][LDSP];

  const int lane = threadIdx.x & 31;
  const int wid  = threadIdx.x >> 5;
  const int wm = wid >> 2, wn = wid & 3;
  const int lh = lane >> 4, l16 = lane & 15;

  FragF acc[4][2];
#pragma unroll
  for (int i = 0; i < 4; ++i)
#pragma unroll
    for (int j = 0; j < 2; ++j)
#pragma unroll
      for (int r = 0; r < 8; ++r) acc[i][j].f[r] = 0.0f;

  gemm_core(Ab, W, As, Bs, m0, n0, acc);

#pragma unroll
  for (int fm = 0; fm < 4; ++fm)
#pragma unroll
    for (int fn = 0; fn < 2; ++fn)
#pragma unroll
      for (int r = 0; r < 8; ++r) {
        const int m = m0 + wm * 64 + fm * 16 + r + 8 * lh;
        const int n = n0 + wn * 32 + fn * 16 + l16;
        out[(size_t)m * D_MODEL + n] = acc[fm][fn].f[r];
      }
}

// ---------------------------------------------------------------------------
// Sliding-window flash attention.
// Block = 4 waves = 64 q rows of one (b,h). All waves loop the union window
// [Q0-255, Q0+63] on K/V tiles staged in LDS via async copies (double
// buffered). Per-wave 16-row online softmax; masked probs forced to 0.
// ---------------------------------------------------------------------------
__launch_bounds__(128)
__global__ void attn_kernel(const unsigned short* __restrict__ Qb,   // [B,H,S,64]
                            const unsigned short* __restrict__ Kb,   // [B,H,S,64]
                            const unsigned short* __restrict__ Vt,   // [B,H,64,S]
                            const int* __restrict__ amask,           // [B,S]
                            unsigned short* __restrict__ Ob) {       // [B,S,1024]
  const int tid  = threadIdx.x;
  const int lane = tid & 31;
  const int wid  = tid >> 5;   // 0..3
  const int lh   = lane >> 4;
  const int l16  = lane & 15;

  const int bh = blockIdx.y;              // b*H + h
  const int b  = bh >> 4;
  const int h  = bh & 15;
  const int Q0 = blockIdx.x * 64;         // block q base
  const int q0 = Q0 + wid * 16;           // wave q tile

  const unsigned short* __restrict__ Qp = Qb + (size_t)bh * SEQ * HEAD_DIM;
  const unsigned short* __restrict__ Kp = Kb + (size_t)bh * SEQ * HEAD_DIM;
  const unsigned short* __restrict__ Vp = Vt + (size_t)bh * HEAD_DIM * SEQ;

  __shared__ __align__(16) unsigned short Ks[2][32][72];   // [key][d], padded
  __shared__ __align__(16) unsigned short Vs[2][64][LDSP]; // [d][key], padded
  __shared__ __align__(16) unsigned short Ps[4][16][LDSP]; // wave-private P

  // Q fragments (d=64 -> two 16x32 A-frags), kept in registers
  FragBF qf[2];
#pragma unroll
  for (int kf = 0; kf < 2; ++kf)
#pragma unroll
    for (int j = 0; j < 8; ++j) {
      const int kk = kf * 32 + 16 * lh + 2 * j;
      qf[kf].u[j] =
          *(const unsigned int*)(Qp + (size_t)(q0 + l16) * HEAD_DIM + kk);
    }

  float mrow[8], lrow[8];
  FragF o[4];
#pragma unroll
  for (int r = 0; r < 8; ++r) { mrow[r] = -1e30f; lrow[r] = 0.0f; }
#pragma unroll
  for (int fd = 0; fd < 4; ++fd)
#pragma unroll
    for (int r = 0; r < 8; ++r) o[fd].f[r] = 0.0f;

  int lo = Q0 - (WINDOW - 1);
  if (lo < 0) lo = 0;
  const int kt0 = (lo >> 5) << 5;
  const int ntiles = ((Q0 + 63 - kt0) >> 5) + 1;   // last tile ends at Q0+63
  const float sc = 0.125f;                         // 1/sqrt(64)

  // staging split: 128 threads; K tile 32x64 (4 thr/row), V tile 64x32 (2/row)
  const int krow = tid >> 2, kcol = (tid & 3) * 16;
  const int vrow = tid >> 1, vcol = (tid & 1) * 16;

  // prologue: stage tile 0 into buffer 0
  async_copy_b128(&Ks[0][krow][kcol], Kp + (size_t)(kt0 + krow) * HEAD_DIM + kcol);
  async_copy_b128(&Vs[0][vrow][vcol], Vp + (size_t)vrow * SEQ + kt0 + vcol);

  for (int t = 0; t < ntiles; ++t) {
    const int cur = t & 1;
    const int kt  = kt0 + t * 32;
    wait_async0();
    __syncthreads();
    if (t + 1 < ntiles) {
      const int kn = kt + 32;
      async_copy_b128(&Ks[cur ^ 1][krow][kcol],
                      Kp + (size_t)(kn + krow) * HEAD_DIM + kcol);
      async_copy_b128(&Vs[cur ^ 1][vrow][vcol],
                      Vp + (size_t)vrow * SEQ + kn + vcol);
    }

    // --- scores: S = Q * K^T over this 32-key tile (from LDS) ---
    FragF s[2];
#pragma unroll
    for (int fn = 0; fn < 2; ++fn) {
      FragBF b0v, b1v;
      const int kr = fn * 16 + l16;
#pragma unroll
      for (int j = 0; j < 8; ++j) {
        b0v.u[j] = *(const unsigned int*)&Ks[cur][kr][16 * lh + 2 * j];
        b1v.u[j] = *(const unsigned int*)&Ks[cur][kr][32 + 16 * lh + 2 * j];
      }
      v8f z;
#pragma unroll
      for (int r = 0; r < 8; ++r) z[r] = 0.0f;
      z = __builtin_amdgcn_wmma_f32_16x16x32_bf16(false, qf[0].v, false, b0v.v,
                                                  (short)0, z, false, false);
      z = __builtin_amdgcn_wmma_f32_16x16x32_bf16(false, qf[1].v, false, b1v.v,
                                                  (short)0, z, false, false);
      s[fn].v = z;
    }

    // --- scale + sliding-window mask + attention_mask ---
#pragma unroll
    for (int fn = 0; fn < 2; ++fn) {
      const int key = kt + fn * 16 + l16;
      const bool mk = (amask[b * SEQ + key] != 0);
#pragma unroll
      for (int r = 0; r < 8; ++r) {
        const int qrow = q0 + r + 8 * lh;
        const bool keep = mk && (key <= qrow) && ((qrow - key) < WINDOW);
        s[fn].f[r] = keep ? s[fn].f[r] * sc : -1e30f;
      }
    }

    // --- online softmax (each row lives across a 16-lane half) ---
    float rmax[8];
#pragma unroll
    for (int r = 0; r < 8; ++r) rmax[r] = fmaxf(s[0].f[r], s[1].f[r]);
#pragma unroll
    for (int off = 1; off <= 8; off <<= 1)
#pragma unroll
      for (int r = 0; r < 8; ++r)
        rmax[r] = fmaxf(rmax[r], __shfl_xor(rmax[r], off, 32));

    float alpha[8];
#pragma unroll
    for (int r = 0; r < 8; ++r) {
      const float mn = fmaxf(mrow[r], rmax[r]);
      alpha[r] = __expf(mrow[r] - mn);
      mrow[r] = mn;
    }
    float rsum[8];
#pragma unroll
    for (int r = 0; r < 8; ++r) rsum[r] = 0.0f;
#pragma unroll
    for (int fn = 0; fn < 2; ++fn)
#pragma unroll
      for (int r = 0; r < 8; ++r) {
        // masked entries contribute exactly 0 (robust for fully-masked tiles)
        const float p =
            (s[fn].f[r] > -1e29f) ? __expf(s[fn].f[r] - mrow[r]) : 0.0f;
        s[fn].f[r] = p;
        rsum[r] += p;
      }
#pragma unroll
    for (int off = 1; off <= 8; off <<= 1)
#pragma unroll
      for (int r = 0; r < 8; ++r) rsum[r] += __shfl_xor(rsum[r], off, 32);
#pragma unroll
    for (int r = 0; r < 8; ++r) lrow[r] = lrow[r] * alpha[r] + rsum[r];
#pragma unroll
    for (int fd = 0; fd < 4; ++fd)
#pragma unroll
      for (int r = 0; r < 8; ++r) o[fd].f[r] *= alpha[r];

    // --- reshape P (C-layout -> A-layout) through wave-private LDS ---
    asm volatile("s_wait_dscnt 0x0" ::: "memory");
#pragma unroll
    for (int fn = 0; fn < 2; ++fn)
#pragma unroll
      for (int r = 0; r < 8; ++r)
        Ps[wid][r + 8 * lh][fn * 16 + l16] = f32_to_bf16(s[fn].f[r]);
    asm volatile("s_wait_dscnt 0x0" ::: "memory");

    FragBF pf;
#pragma unroll
    for (int j = 0; j < 8; ++j) {
      const int kk = 16 * lh + 2 * j;
      pf.u[j] = *(const unsigned int*)&Ps[wid][l16][kk];
    }

    // --- O += P * V (Vs rows are d, contiguous along keys) ---
#pragma unroll
    for (int fd = 0; fd < 4; ++fd) {
      FragBF vb;
      const int vr = fd * 16 + l16;
#pragma unroll
      for (int j = 0; j < 8; ++j)
        vb.u[j] = *(const unsigned int*)&Vs[cur][vr][16 * lh + 2 * j];
      o[fd].v = __builtin_amdgcn_wmma_f32_16x16x32_bf16(
          false, pf.v, false, vb.v, (short)0, o[fd].v, false, false);
    }
    __syncthreads();   // all reads of buffer `cur` done before restage
  }

  // --- normalize + store bf16 [B,S,1024] for the Wo GEMM ---
  float invl[8];
#pragma unroll
  for (int r = 0; r < 8; ++r) invl[r] = 1.0f / lrow[r];
#pragma unroll
  for (int fd = 0; fd < 4; ++fd)
#pragma unroll
    for (int r = 0; r < 8; ++r) {
      const int qrow = q0 + r + 8 * lh;
      Ob[((size_t)b * SEQ + qrow) * D_MODEL + h * HEAD_DIM + fd * 16 + l16] =
          f32_to_bf16(o[fd].f[r] * invl[r]);
    }
}

// ---------------------------------------------------------------------------
// Host launcher
// ---------------------------------------------------------------------------
extern "C" void kernel_launch(void* const* d_in, const int* in_sizes, int n_in,
                              void* d_out, int out_size, void* d_ws,
                              size_t ws_size, hipStream_t stream) {
  (void)in_sizes; (void)n_in; (void)out_size; (void)ws_size;
  const float* X  = (const float*)d_in[0];
  const int*   am = (const int*)d_in[1];
  const float* Wq = (const float*)d_in[2];
  const float* Wk = (const float*)d_in[3];
  const float* Wv = (const float*)d_in[4];
  const float* Wo = (const float*)d_in[5];
  float* out = (float*)d_out;

  unsigned short* ws = (unsigned short*)d_ws;
  size_t off = 0;
  auto take = [&](size_t n) { unsigned short* p = ws + off; off += n; return p; };
  unsigned short* Xb  = take((size_t)M_TOTAL * D_MODEL);
  unsigned short* Wqb = take((size_t)D_MODEL * D_MODEL);
  unsigned short* Wkb = take((size_t)D_MODEL * D_MODEL);
  unsigned short* Wvb = take((size_t)D_MODEL * D_MODEL);
  unsigned short* Wob = take((size_t)D_MODEL * D_MODEL);
  unsigned short* Qb  = take((size_t)BATCH * NUM_HEADS * SEQ * HEAD_DIM);
  unsigned short* Kb  = take((size_t)BATCH * NUM_HEADS * SEQ * HEAD_DIM);
  unsigned short* Vtb = take((size_t)BATCH * NUM_HEADS * SEQ * HEAD_DIM);
  unsigned short* Ab  = take((size_t)M_TOTAL * D_MODEL);

  const int nX = M_TOTAL * D_MODEL;   // 4 Mi elems
  const int nW = D_MODEL * D_MODEL;   // 1 Mi elems
  cvt_f32_bf16<<<nX / 8 / 256, 256, 0, stream>>>(X, Xb, nX);
  cvt_f32_bf16<<<nW / 8 / 256, 256, 0, stream>>>(Wq, Wqb, nW);
  cvt_f32_bf16<<<nW / 8 / 256, 256, 0, stream>>>(Wk, Wkb, nW);
  cvt_f32_bf16<<<nW / 8 / 256, 256, 0, stream>>>(Wv, Wvb, nW);
  cvt_f32_bf16<<<nW / 8 / 256, 256, 0, stream>>>(Wo, Wob, nW);

  gemm_qkv_kernel<<<dim3(D_MODEL / BN, M_TOTAL / BM, 3), 256, 0, stream>>>(
      Xb, Wqb, Wkb, Wvb, Qb, Kb, Vtb);

  attn_kernel<<<dim3(SEQ / 64, BATCH * NUM_HEADS), 128, 0, stream>>>(
      Qb, Kb, Vtb, am, Ab);

  gemm_out_kernel<<<dim3(D_MODEL / BN, M_TOTAL / BM), 256, 0, stream>>>(
      Ab, Wob, out);
}
